// GGNNLayer_37426345017664
// MI455X (gfx1250) — matrix operand order repaired and checked by reading, not compiled, not run on hardware
//
#include <hip/hip_runtime.h>
#include <hip/hip_bf16.h>

typedef __attribute__((ext_vector_type(16))) __bf16 bf16x16;
typedef __attribute__((ext_vector_type(8)))  __bf16 bf16x8;
typedef __attribute__((ext_vector_type(8)))  float  f32x8;

#define B_    256
#define N_    512
#define DIN   300
#define D_    96
#define R_    (B_ * N_)        // 131072 rows
#define NT    6                // D/16 col tiles
#define KT_D  3                // D/32 k tiles
#define KT_ENC 10              // ceil(300/32)
#define KT_N  16               // N/32 k tiles

// swizzled weight sizes (in halfs)
#define SZ_WENC  (KT_ENC * NT * 512)       // 30720
#define SZ_GATE  (KT_D * NT * 512)         // 9216
#define SZ_WSW_H (SZ_WENC + 6 * SZ_GATE)   // 86016

// workspace byte offsets
#define OFF_WSW   0
#define OFF_XF32  (172032)                       // 86016*2 rounded
#define OFF_AF32  (OFF_XF32 + (size_t)R_ * D_ * 4)
#define OFF_XRM   (OFF_AF32 + (size_t)R_ * D_ * 4)
#define OFF_XBF   (OFF_XRM + (size_t)R_ * D_ * 2)

__device__ __forceinline__ f32x8 zero8() {
  f32x8 z;
#pragma unroll
  for (int i = 0; i < 8; ++i) z[i] = 0.0f;
  return z;
}

__device__ __forceinline__ f32x8 wmma_bf16(bf16x16 a, bf16x16 b, f32x8 c) {
  return __builtin_amdgcn_wmma_f32_16x16x32_bf16(false, a, false, b, (short)0, c,
                                                 false, false);
}

// pack 8 contiguous fp32 into bf16 fragment elements [base..base+7]
__device__ __forceinline__ void pack8(bf16x16& a, int base, const float* p) {
  float4 q0 = *(const float4*)p;
  float4 q1 = *(const float4*)(p + 4);
  a[base + 0] = (__bf16)q0.x; a[base + 1] = (__bf16)q0.y;
  a[base + 2] = (__bf16)q0.z; a[base + 3] = (__bf16)q0.w;
  a[base + 4] = (__bf16)q1.x; a[base + 5] = (__bf16)q1.y;
  a[base + 6] = (__bf16)q1.z; a[base + 7] = (__bf16)q1.w;
}

__device__ __forceinline__ void pack8_relu_mask(bf16x16& a, int base, const float* p,
                                                float mk) {
  float4 q0 = *(const float4*)p;
  float4 q1 = *(const float4*)(p + 4);
  a[base + 0] = (__bf16)(fmaxf(q0.x, 0.0f) * mk);
  a[base + 1] = (__bf16)(fmaxf(q0.y, 0.0f) * mk);
  a[base + 2] = (__bf16)(fmaxf(q0.z, 0.0f) * mk);
  a[base + 3] = (__bf16)(fmaxf(q0.w, 0.0f) * mk);
  a[base + 4] = (__bf16)(fmaxf(q1.x, 0.0f) * mk);
  a[base + 5] = (__bf16)(fmaxf(q1.y, 0.0f) * mk);
  a[base + 6] = (__bf16)(fmaxf(q1.z, 0.0f) * mk);
  a[base + 7] = (__bf16)(fmaxf(q1.w, 0.0f) * mk);
}

// ---------------------------------------------------------------------------
// Weight prep: fp32 -> bf16, pre-swizzled to WMMA B-fragment layout:
//   frag[kt][nt][lane][e] = W[kt*32 + (e<8?e:e+8) + (lane>=16?8:0)][nt*16 + (lane&15)]
// ---------------------------------------------------------------------------
__global__ void k_prep_weights(const float* __restrict__ Wenc,
                               const float* __restrict__ Wz0, const float* __restrict__ Wz1,
                               const float* __restrict__ Wr0, const float* __restrict__ Wr1,
                               const float* __restrict__ Wh0, const float* __restrict__ Wh1,
                               __bf16* __restrict__ dst) {
  int idx = blockIdx.x * blockDim.x + threadIdx.x;
  if (idx >= SZ_WSW_H) return;
  if (idx < SZ_WENC) {
    int kt = idx / (NT * 512);
    int r = idx % (NT * 512);
    int nt = r / 512;
    int lane = (r % 512) / 16;
    int e = r % 16;
    int k = kt * 32 + (e < 8 ? e : e + 8) + ((lane >= 16) ? 8 : 0);
    int n = nt * 16 + (lane & 15);
    float v = (k < DIN) ? Wenc[k * D_ + n] : 0.0f;
    dst[idx] = (__bf16)v;
  } else {
    int i2 = idx - SZ_WENC;
    int w = i2 / SZ_GATE;
    int r = i2 % SZ_GATE;
    int kt = r / (NT * 512);
    int r2 = r % (NT * 512);
    int nt = r2 / 512;
    int lane = (r2 % 512) / 16;
    int e = r2 % 16;
    int k = kt * 32 + (e < 8 ? e : e + 8) + ((lane >= 16) ? 8 : 0);
    int n = nt * 16 + (lane & 15);
    const float* W = (w == 0) ? Wz0 : (w == 1) ? Wz1 : (w == 2) ? Wr0
                   : (w == 3) ? Wr1 : (w == 4) ? Wh0 : Wh1;
    dst[idx] = (__bf16)W[k * D_ + n];
  }
}

// ---------------------------------------------------------------------------
// Encoder: out = (mask * relu(x)) @ W_enc + b_enc       [R,300]x[300,96]
// One wave per 16-row tile. K padded 300 -> 320 with zeros.
// ---------------------------------------------------------------------------
__global__ void k_encoder(const float* __restrict__ x, const float* __restrict__ mask,
                          const __bf16* __restrict__ wsw, const float* __restrict__ benc,
                          float* __restrict__ xf, __bf16* __restrict__ xrm) {
  const int lane = threadIdx.x & 31;
  const int tile = (blockIdx.x * blockDim.x + threadIdx.x) >> 5;  // 0..8191
  const int row0 = tile * 16;
  const int m = lane & 15;
  const int hi = lane >> 4;
  const float* rowp = x + (size_t)(row0 + m) * DIN;
  const float mk = mask[row0 + m];

  f32x8 acc[NT];
#pragma unroll
  for (int nt = 0; nt < NT; ++nt) acc[nt] = zero8();

  for (int kt = 0; kt < KT_ENC; ++kt) {
    const int krun0 = kt * 32 + hi * 8;
    bf16x16 a;
    if (kt < 9) {
      pack8_relu_mask(a, 0, rowp + krun0, mk);
      pack8_relu_mask(a, 8, rowp + krun0 + 16, mk);
    } else {
#pragma unroll
      for (int e = 0; e < 16; ++e) {
        int k = krun0 + (e < 8 ? e : e + 8);
        float v = (k < DIN) ? rowp[k] : 0.0f;
        a[e] = (__bf16)(fmaxf(v, 0.0f) * mk);
      }
    }
    const __bf16* wb = wsw + kt * (NT * 512);
#pragma unroll
    for (int nt = 0; nt < NT; ++nt) {
      bf16x16 b = *(const bf16x16*)(wb + nt * 512 + lane * 16);
      acc[nt] = wmma_bf16(a, b, acc[nt]);
    }
  }

#pragma unroll
  for (int nt = 0; nt < NT; ++nt) {
    int n = nt * 16 + m;
    float bias = benc[n];
#pragma unroll
    for (int i = 0; i < 8; ++i) {
      size_t row = row0 + i + hi * 8;
      float v = acc[nt][i] + bias;
      xf[row * D_ + n] = v;
      xrm[row * D_ + n] = (__bf16)v;
    }
  }
}

// ---------------------------------------------------------------------------
// Swizzle: row-major bf16 state [R,96] -> B-fragment layout
// One wave per (batch, 32-row k-tile).
// ---------------------------------------------------------------------------
__global__ void k_swizzle(const __bf16* __restrict__ xrm, __bf16* __restrict__ xbf) {
  __shared__ __bf16 lds[8][32 * D_];
  const int lane = threadIdx.x & 31;
  const int wv = threadIdx.x >> 5;
  const int gw = (blockIdx.x * blockDim.x + threadIdx.x) >> 5;  // 0..4095 = b*16+kt

  const uint32_t* src = (const uint32_t*)(xrm + (size_t)gw * 32 * D_);
  uint32_t* l32 = (uint32_t*)lds[wv];
  for (int t = lane; t < (32 * D_) / 2; t += 32) l32[t] = src[t];
  __syncthreads();

  __bf16* dst = xbf + (size_t)gw * (NT * 512);
#pragma unroll
  for (int nt = 0; nt < NT; ++nt) {
    bf16x16 v;
#pragma unroll
    for (int e = 0; e < 16; ++e) {
      int k = (e < 8 ? e : e + 8) + ((lane >= 16) ? 8 : 0);
      int n = nt * 16 + (lane & 15);
      v[e] = lds[wv][k * D_ + n];
    }
    *(bf16x16*)(dst + nt * 512 + lane * 16) = v;
  }
}

// ---------------------------------------------------------------------------
// Aggregation: a = adj @ x          per batch [512,512]x[512,96]
// One wave per (batch, 16-row tile). adj fp32 converted to bf16 in-register.
// ---------------------------------------------------------------------------
__global__ void k_agg(const float* __restrict__ adj, const __bf16* __restrict__ xbf,
                      float* __restrict__ ao) {
  const int lane = threadIdx.x & 31;
  const int gw = (blockIdx.x * blockDim.x + threadIdx.x) >> 5;  // 0..8191
  const int b = gw >> 5;
  const int rt = gw & 31;
  const int m = lane & 15;
  const int hi = lane >> 4;

  const float* arow = adj + ((size_t)b * N_ + rt * 16 + m) * N_;
  const __bf16* xb = xbf + (size_t)b * (KT_N * NT * 512);

  f32x8 acc[NT];
#pragma unroll
  for (int nt = 0; nt < NT; ++nt) acc[nt] = zero8();

  for (int kt = 0; kt < KT_N; ++kt) {
    const int krun0 = kt * 32 + hi * 8;
    bf16x16 a;
    pack8(a, 0, arow + krun0);
    pack8(a, 8, arow + krun0 + 16);
    const __bf16* bb = xb + kt * (NT * 512);
#pragma unroll
    for (int nt = 0; nt < NT; ++nt) {
      bf16x16 bf = *(const bf16x16*)(bb + nt * 512 + lane * 16);
      acc[nt] = wmma_bf16(a, bf, acc[nt]);
    }
  }

  const size_t rowbase = (size_t)b * N_ + rt * 16;
#pragma unroll
  for (int nt = 0; nt < NT; ++nt) {
#pragma unroll
    for (int i = 0; i < 8; ++i) {
      ao[(rowbase + i + hi * 8) * D_ + nt * 16 + m] = acc[nt][i];
    }
  }
}

// ---------------------------------------------------------------------------
// Gates: z = sig(a@Wz0 + x@Wz1 + bz), r = sig(a@Wr0 + x@Wr1 + br),
//        h = relu(mask*(a@Wh0 + (r*x)@Wh1 + bh)), out = h*z + x*(1-z)
// One wave per 16-row tile. r*x bounced through LDS to become an A-fragment.
// xf and of may alias (in-place step 1) -> no restrict on them.
// ---------------------------------------------------------------------------
__global__ void k_gates(const float* __restrict__ ag, const float* xf,
                        const float* __restrict__ mask, const __bf16* __restrict__ gw,
                        const float* __restrict__ bz0, const float* __restrict__ bz1,
                        const float* __restrict__ br0, const float* __restrict__ br1,
                        const float* __restrict__ bh0, const float* __restrict__ bh1,
                        float* of, __bf16* __restrict__ xrm) {
  __shared__ __bf16 lds[8][16 * D_];
  const int lane = threadIdx.x & 31;
  const int wv = threadIdx.x >> 5;
  const int tile = (blockIdx.x * blockDim.x + threadIdx.x) >> 5;  // 0..8191
  const int row0 = tile * 16;
  const int m = lane & 15;
  const int hi = lane >> 4;

  const float* arow = ag + (size_t)(row0 + m) * D_;
  const float* xrow = xf + (size_t)(row0 + m) * D_;
  const __bf16* Wz0f = gw;
  const __bf16* Wz1f = gw + 1 * SZ_GATE;
  const __bf16* Wr0f = gw + 2 * SZ_GATE;
  const __bf16* Wr1f = gw + 3 * SZ_GATE;
  const __bf16* Wh0f = gw + 4 * SZ_GATE;
  const __bf16* Wh1f = gw + 5 * SZ_GATE;

  f32x8 accz[NT], accr[NT], acch[NT];
#pragma unroll
  for (int nt = 0; nt < NT; ++nt) { accz[nt] = zero8(); accr[nt] = zero8(); acch[nt] = zero8(); }

  for (int kt = 0; kt < KT_D; ++kt) {
    const int krun0 = kt * 32 + hi * 8;
    bf16x16 aA, aX;
    pack8(aA, 0, arow + krun0); pack8(aA, 8, arow + krun0 + 16);
    pack8(aX, 0, xrow + krun0); pack8(aX, 8, xrow + krun0 + 16);
    const int kb = kt * (NT * 512);
#pragma unroll
    for (int nt = 0; nt < NT; ++nt) {
      const int o = kb + nt * 512 + lane * 16;
      accz[nt] = wmma_bf16(aA, *(const bf16x16*)(Wz0f + o), accz[nt]);
      accz[nt] = wmma_bf16(aX, *(const bf16x16*)(Wz1f + o), accz[nt]);
      accr[nt] = wmma_bf16(aA, *(const bf16x16*)(Wr0f + o), accr[nt]);
      accr[nt] = wmma_bf16(aX, *(const bf16x16*)(Wr1f + o), accr[nt]);
      acch[nt] = wmma_bf16(aA, *(const bf16x16*)(Wh0f + o), acch[nt]);
    }
  }

  // z kept in accz; r*x staged into LDS (row-major bf16 16x96 tile)
#pragma unroll
  for (int nt = 0; nt < NT; ++nt) {
    int n = nt * 16 + m;
    float bzv = bz0[n] + bz1[n];
    float brv = br0[n] + br1[n];
#pragma unroll
    for (int i = 0; i < 8; ++i) {
      int rm = i + hi * 8;
      float zv = 1.0f / (1.0f + __expf(-(accz[nt][i] + bzv)));
      accz[nt][i] = zv;
      float rv = 1.0f / (1.0f + __expf(-(accr[nt][i] + brv)));
      float xv = xf[(size_t)(row0 + rm) * D_ + n];
      lds[wv][rm * D_ + n] = (__bf16)(rv * xv);
    }
  }
  __syncthreads();

  for (int kt = 0; kt < KT_D; ++kt) {
    const int krun0 = kt * 32 + hi * 8;
    bf16x8 r0 = *(const bf16x8*)&lds[wv][m * D_ + krun0];
    bf16x8 r1 = *(const bf16x8*)&lds[wv][m * D_ + krun0 + 16];
    bf16x16 aR;
#pragma unroll
    for (int e = 0; e < 8; ++e) { aR[e] = r0[e]; aR[e + 8] = r1[e]; }
    const int kb = kt * (NT * 512);
#pragma unroll
    for (int nt = 0; nt < NT; ++nt)
      acch[nt] = wmma_bf16(aR, *(const bf16x16*)(Wh1f + kb + nt * 512 + lane * 16),
                           acch[nt]);
  }

#pragma unroll
  for (int nt = 0; nt < NT; ++nt) {
    int n = nt * 16 + m;
    float bhv = bh0[n] + bh1[n];
#pragma unroll
    for (int i = 0; i < 8; ++i) {
      int rm = i + hi * 8;
      size_t row = row0 + rm;
      float mk = mask[row];
      float hv = fmaxf(mk * (acch[nt][i] + bhv), 0.0f);
      float zv = accz[nt][i];
      float xv = xf[row * D_ + n];
      float ov = hv * zv + xv * (1.0f - zv);
      of[row * D_ + n] = ov;
      xrm[row * D_ + n] = (__bf16)ov;
    }
  }
}

// ---------------------------------------------------------------------------
extern "C" void kernel_launch(void* const* d_in, const int* in_sizes, int n_in,
                              void* d_out, int out_size, void* d_ws, size_t ws_size,
                              hipStream_t stream) {
  const float* x    = (const float*)d_in[0];
  const float* adj  = (const float*)d_in[1];
  const float* mask = (const float*)d_in[2];
  const float* Wenc = (const float*)d_in[3];
  const float* benc = (const float*)d_in[4];
  const float* Wz0  = (const float*)d_in[5];
  const float* Wz1  = (const float*)d_in[6];
  const float* Wr0  = (const float*)d_in[7];
  const float* Wr1  = (const float*)d_in[8];
  const float* Wh0  = (const float*)d_in[9];
  const float* Wh1  = (const float*)d_in[10];
  const float* bz0  = (const float*)d_in[11];
  const float* bz1  = (const float*)d_in[12];
  const float* br0  = (const float*)d_in[13];
  const float* br1  = (const float*)d_in[14];
  const float* bh0  = (const float*)d_in[15];
  const float* bh1  = (const float*)d_in[16];

  char* ws = (char*)d_ws;
  __bf16* wsw = (__bf16*)(ws + OFF_WSW);
  float*  xf  = (float*)(ws + OFF_XF32);
  float*  af  = (float*)(ws + OFF_AF32);
  __bf16* xrm = (__bf16*)(ws + OFF_XRM);
  __bf16* xbf = (__bf16*)(ws + OFF_XBF);
  float*  out = (float*)d_out;
  const __bf16* gw = wsw + SZ_WENC;

  k_prep_weights<<<(SZ_WSW_H + 255) / 256, 256, 0, stream>>>(Wenc, Wz0, Wz1, Wr0, Wr1,
                                                             Wh0, Wh1, wsw);
  k_encoder<<<1024, 256, 0, stream>>>(x, mask, wsw, benc, xf, xrm);
  k_swizzle<<<512, 256, 0, stream>>>(xrm, xbf);

  // GRU step 1 (state updated in place)
  k_agg<<<1024, 256, 0, stream>>>(adj, xbf, af);
  k_gates<<<1024, 256, 0, stream>>>(af, xf, mask, gw, bz0, bz1, br0, br1, bh0, bh1,
                                    xf, xrm);
  k_swizzle<<<512, 256, 0, stream>>>(xrm, xbf);

  // GRU step 2 (final state -> d_out)
  k_agg<<<1024, 256, 0, stream>>>(adj, xbf, af);
  k_gates<<<1024, 256, 0, stream>>>(af, xf, mask, gw, bz0, bz1, br0, br1, bh0, bh1,
                                    out, xrm);
}
